// EncoderWithLSTM_45921790329229
// MI455X (gfx1250) — matrix operand order repaired and checked by reading, not compiled, not run on hardware
//
#include <hip/hip_runtime.h>

// ---------------------------------------------------------------------------
// 2-layer bidirectional LSTM for MI455X (gfx1250, wave32, WMMA bf16).
// T=160, B=256, C=512, H=256, G=4H=1024.
// ---------------------------------------------------------------------------

typedef __attribute__((ext_vector_type(16))) __bf16 v16bf;
typedef __attribute__((ext_vector_type(8)))  __bf16 v8bf;
typedef __attribute__((ext_vector_type(8)))  float  v8f;

#define TSTEPS 160
#define BATCH  256
#define HID    256
#define GDIM   1024
#define NBLK   32   // persistent recurrent kernel blocks (co-resident)

static __device__ inline v8bf ldb8(const __bf16* p) {
  return *(const v8bf*)p;   // 16B contiguous -> global_load_b128
}

static __device__ inline v16bf mk16(v8bf lo, v8bf hi) {
  return __builtin_shufflevector(lo, hi, 0,1,2,3,4,5,6,7,8,9,10,11,12,13,14,15);
}

static __device__ inline float sigm(float x) {
  return 1.0f / (1.0f + __expf(-x));
}
static __device__ inline float tanh_f(float x) {
  return 1.0f - 2.0f / (__expf(2.0f * x) + 1.0f);
}

// ------------------------- grid-wide generation barrier --------------------
static __device__ inline void gbar(unsigned* cnt, unsigned* gen, unsigned nb) {
  __threadfence();          // make h stores agent-visible before arrive
  __syncthreads();
  if (threadIdx.x == 0) {
    unsigned g = __hip_atomic_load(gen, __ATOMIC_RELAXED, __HIP_MEMORY_SCOPE_AGENT);
    unsigned a = __hip_atomic_fetch_add(cnt, 1u, __ATOMIC_ACQ_REL, __HIP_MEMORY_SCOPE_AGENT);
    if (a == nb - 1u) {
      __hip_atomic_store(cnt, 0u,    __ATOMIC_RELAXED, __HIP_MEMORY_SCOPE_AGENT);
      __hip_atomic_store(gen, g + 1u, __ATOMIC_RELEASE, __HIP_MEMORY_SCOPE_AGENT);
    } else {
      while (__hip_atomic_load(gen, __ATOMIC_ACQUIRE, __HIP_MEMORY_SCOPE_AGENT) == g)
        __builtin_amdgcn_s_sleep(2);
    }
  }
  __syncthreads();
}

// ------------------------------- small utility kernels ---------------------
__global__ void init_bar_k(unsigned* bar) {
  if (threadIdx.x < 2) bar[threadIdx.x] = 0u;
}

__global__ void f32_to_bf16_k(const float* __restrict__ in,
                              __bf16* __restrict__ out, int n) {
  int i = blockIdx.x * blockDim.x + threadIdx.x;
  int stride = gridDim.x * blockDim.x;
  for (; i < n; i += stride) out[i] = (__bf16)in[i];
}

__global__ void bias_comb_k(const float* __restrict__ a,
                            const float* __restrict__ b,
                            float* __restrict__ out, int n) {
  int i = blockIdx.x * blockDim.x + threadIdx.x;
  if (i < n) out[i] = a[i] + b[i];
}

// ---------------------------------------------------------------------------
// xg GEMM: O[m][n] = sum_k A[m][k] * W[n][k] + bias[n]   (both directions)
// A: [40960, Kdim=512] bf16 row-major.  W: [1024, 512] bf16 row-major.
// Output in WMMA-fragment order: [Mtile(2560)][ntile(64)][lane(32)][reg(8)] bf16
// grid (320, 8, 2), block 256 (8 waves); one wave = 16x128 strip, 128 WMMAs.
// ---------------------------------------------------------------------------
__global__ __launch_bounds__(256) void gemm_xg_k(
    const __bf16* __restrict__ A,
    const __bf16* __restrict__ W0, const __bf16* __restrict__ W1,
    const float*  __restrict__ bias0, const float* __restrict__ bias1,
    __bf16* __restrict__ O0, __bf16* __restrict__ O1) {
  const int d = blockIdx.z;
  const __bf16* W   = d ? W1 : W0;
  const float*  bias = d ? bias1 : bias0;
  __bf16*       O   = d ? O1 : O0;

  const int lane = threadIdx.x & 31;
  const int wv   = threadIdx.x >> 5;
  const size_t Mt = (size_t)blockIdx.x * 8 + wv;   // 0..2559
  const int ntb  = blockIdx.y * 8;                 // ntile base (of 64)
  const int half = lane >> 4;
  const int col  = lane & 15;

  const __bf16* arow = A + (Mt * 16 + col) * 512;

  v8f acc[8] = {};
  for (int ks = 0; ks < 16; ++ks) {
    const int k0 = ks * 32;
    // A 16x32 bf16 fragment: lane(col,half) holds K {k0+8h..+7, k0+16+8h..+7}
    v16bf a = mk16(ldb8(arow + k0 + 8 * half),
                   ldb8(arow + k0 + 16 + 8 * half));
#pragma unroll
    for (int j = 0; j < 8; ++j) {
      // B 32x16 fragment: lane(col,g) holds K {k0+8g..+7, k0+16+8g..+7} of row n
      const __bf16* brow = W + ((size_t)((ntb + j) * 16 + col)) * 512 + k0 + 8 * half;
      v16bf b = mk16(ldb8(brow), ldb8(brow + 16));
      acc[j] = __builtin_amdgcn_wmma_f32_16x16x32_bf16(
          false, a, false, b, (short)0, acc[j], false, false);
    }
  }
#pragma unroll
  for (int j = 0; j < 8; ++j) {
    const float bv = bias[(ntb + j) * 16 + col];
    v8bf o;
#pragma unroll
    for (int r = 0; r < 8; ++r) o[r] = (__bf16)(acc[j][r] + bv);
    *(v8bf*)(O + ((Mt * 64 + (ntb + j)) * 32 + lane) * 8) = o;
  }
}

// ---------------------------------------------------------------------------
// Persistent recurrent kernel: both directions of one layer, 160 steps.
// 32 blocks x 256 threads = 256 waves; wave owns tile-set s=0 (fwd) and
// s=1 (bwd): one 16(batch)x16(hidden) tile with all 4 gates.
// Cell state c stays in registers for the entire sequence.
// hbuf: [2 buf][2 dir][256][256] bf16 (double buffered hidden state).
// ---------------------------------------------------------------------------
__global__ __launch_bounds__(256) void lstm_rec_k(
    const __bf16* __restrict__ xg0, const __bf16* __restrict__ xg1,
    const __bf16* __restrict__ whh0, const __bf16* __restrict__ whh1,
    __bf16* __restrict__ hbuf,
    __bf16* __restrict__ out_bf,   // layer-0: [160][256][512] bf16 (else null)
    float*  __restrict__ out_f32,  // layer-1: [160][256][512] f32  (else null)
    unsigned* __restrict__ bar) {
  const int tid  = threadIdx.x;
  const int lane = tid & 31;
  const int wave = (blockIdx.x << 3) | (tid >> 5);  // 0..255
  const int half = lane >> 4;
  const int col  = lane & 15;

  // zero both hidden-state buffers (h_0 = 0); 131072 dwords
  {
    unsigned* hz = (unsigned*)hbuf;
    for (int i = blockIdx.x * 256 + tid; i < 131072; i += NBLK * 256) hz[i] = 0u;
  }
  gbar(bar, bar + 1, NBLK);

  v8f c0 = {};  // cell state, forward tile
  v8f c1 = {};  // cell state, backward tile

  for (int t = 0; t < TSTEPS; ++t) {
#pragma unroll
    for (int s = 0; s < 2; ++s) {
      const int d  = s;                 // s=0 -> forward, s=1 -> backward
      const int mt = wave >> 4;         // batch tile   0..15
      const int ht = wave & 15;         // hidden tile  0..15
      const int tt = d ? (TSTEPS - 1 - t) : t;

      const __bf16* cur = hbuf + ((t & 1) ? 131072 : 0) + d * 65536;
      __bf16*       nxt = hbuf + ((t & 1) ? 0 : 131072) + d * 65536;
      const __bf16* W   = d ? whh1 : whh0;
      const __bf16* XG  = d ? xg1  : xg0;

      const __bf16* arow = cur + (mt * 16 + col) * 256;

      v8f acc[4] = {};
#pragma unroll
      for (int ks = 0; ks < 8; ++ks) {
        const int k0 = ks * 32;
        v16bf a = mk16(ldb8(arow + k0 + 8 * half),
                       ldb8(arow + k0 + 16 + 8 * half));
#pragma unroll
        for (int gate = 0; gate < 4; ++gate) {
          const __bf16* brow =
              W + (size_t)(gate * 256 + ht * 16 + col) * 256 + k0 + 8 * half;
          v16bf b = mk16(ldb8(brow), ldb8(brow + 16));
          acc[gate] = __builtin_amdgcn_wmma_f32_16x16x32_bf16(
              false, a, false, b, (short)0, acc[gate], false, false);
        }
      }

      // add precomputed input gates (bias already folded) — fragment layout
      const size_t Mt = (size_t)tt * 16 + mt;
#pragma unroll
      for (int gate = 0; gate < 4; ++gate) {
        const int nt = gate * 16 + ht;
        v8bf xv = ldb8(XG + ((Mt * 64 + nt) * 32 + lane) * 8);
#pragma unroll
        for (int r = 0; r < 8; ++r) acc[gate][r] += (float)xv[r];
      }

      // prefetch next timestep's xg fragment into the caches
      if (t + 1 < TSTEPS) {
        const int ttn = d ? (TSTEPS - 2 - t) : (t + 1);
        __builtin_prefetch(
            XG + ((((size_t)ttn * 16 + mt) * 64 + ht) * 32 + lane) * 8, 0, 1);
      }

      v8f& c = s ? c1 : c0;
      const int nloc = ht * 16 + col;
#pragma unroll
      for (int r = 0; r < 8; ++r) {
        const float iv = sigm(acc[0][r]);
        const float fv = sigm(acc[1][r]);
        const float gv = tanh_f(acc[2][r]);
        const float ov = sigm(acc[3][r]);
        const float cc = fv * c[r] + iv * gv;
        c[r] = cc;
        const float hv = ov * tanh_f(cc);
        const int mrow = mt * 16 + half * 8 + r;   // C-layout: M = 8*half + reg
        nxt[mrow * 256 + nloc] = (__bf16)hv;
        const size_t oidx = ((size_t)tt * BATCH + mrow) * 512 + d * HID + nloc;
        if (out_f32) out_f32[oidx] = hv;
        else         out_bf[oidx]  = (__bf16)hv;
      }
    }
    gbar(bar, bar + 1, NBLK);
  }
}

// ---------------------------------------------------------------------------
// Host-side launcher. Workspace layout (bytes, ~258.5 MB total):
// ---------------------------------------------------------------------------
extern "C" void kernel_launch(void* const* d_in, const int* in_sizes, int n_in,
                              void* d_out, int out_size, void* d_ws, size_t ws_size,
                              hipStream_t stream) {
  (void)in_sizes; (void)n_in; (void)out_size; (void)ws_size;

  const float* x        = (const float*)d_in[0];
  const float* wih_l0f  = (const float*)d_in[1];
  const float* whh_l0f  = (const float*)d_in[2];
  const float* bih_l0f  = (const float*)d_in[3];
  const float* bhh_l0f  = (const float*)d_in[4];
  const float* wih_l0b  = (const float*)d_in[5];
  const float* whh_l0b  = (const float*)d_in[6];
  const float* bih_l0b  = (const float*)d_in[7];
  const float* bhh_l0b  = (const float*)d_in[8];
  const float* wih_l1f  = (const float*)d_in[9];
  const float* whh_l1f  = (const float*)d_in[10];
  const float* bih_l1f  = (const float*)d_in[11];
  const float* bhh_l1f  = (const float*)d_in[12];
  const float* wih_l1b  = (const float*)d_in[13];
  const float* whh_l1b  = (const float*)d_in[14];
  const float* bih_l1b  = (const float*)d_in[15];
  const float* bhh_l1b  = (const float*)d_in[16];

  char* ws = (char*)d_ws;
  constexpr size_t XBF_E  = (size_t)TSTEPS * BATCH * 512;   // 20,971,520
  constexpr size_t XG_E   = (size_t)TSTEPS * BATCH * GDIM;  // 41,943,040
  constexpr size_t WIH_E  = (size_t)GDIM * 512;             // 524,288
  constexpr size_t WHH_E  = (size_t)GDIM * HID;             // 262,144

  __bf16* xbf   = (__bf16*)(ws + 0);
  __bf16* h0bf  = (__bf16*)(ws + 41943040ull);
  __bf16* xg0   = (__bf16*)(ws + 83886080ull);
  __bf16* xg1   = (__bf16*)(ws + 167772160ull);
  char*   wb    = ws + 251658240ull;
  __bf16* bwih_l0f = (__bf16*)(wb + 0);
  __bf16* bwih_l0b = (__bf16*)(wb + 1048576ull);
  __bf16* bwhh_l0f = (__bf16*)(wb + 2097152ull);
  __bf16* bwhh_l0b = (__bf16*)(wb + 2621440ull);
  __bf16* bwih_l1f = (__bf16*)(wb + 3145728ull);
  __bf16* bwih_l1b = (__bf16*)(wb + 4194304ull);
  __bf16* bwhh_l1f = (__bf16*)(wb + 5242880ull);
  __bf16* bwhh_l1b = (__bf16*)(wb + 5767168ull);
  float* bias_l0f = (float*)(ws + 257949696ull);
  float* bias_l0b = bias_l0f + GDIM;
  float* bias_l1f = bias_l0b + GDIM;
  float* bias_l1b = bias_l1f + GDIM;
  __bf16*  hbuf = (__bf16*)(ws + 257966080ull);  // [2][2][256][256]
  unsigned* bar = (unsigned*)(ws + 258490368ull);

  // 0) barrier init (ws is poisoned; must be zeroed every call)
  init_bar_k<<<1, 64, 0, stream>>>(bar);

  // 1) bf16 conversions
  f32_to_bf16_k<<<4096, 256, 0, stream>>>(x, xbf, (int)XBF_E);
  f32_to_bf16_k<<<512, 256, 0, stream>>>(wih_l0f, bwih_l0f, (int)WIH_E);
  f32_to_bf16_k<<<512, 256, 0, stream>>>(wih_l0b, bwih_l0b, (int)WIH_E);
  f32_to_bf16_k<<<256, 256, 0, stream>>>(whh_l0f, bwhh_l0f, (int)WHH_E);
  f32_to_bf16_k<<<256, 256, 0, stream>>>(whh_l0b, bwhh_l0b, (int)WHH_E);
  f32_to_bf16_k<<<512, 256, 0, stream>>>(wih_l1f, bwih_l1f, (int)WIH_E);
  f32_to_bf16_k<<<512, 256, 0, stream>>>(wih_l1b, bwih_l1b, (int)WIH_E);
  f32_to_bf16_k<<<256, 256, 0, stream>>>(whh_l1f, bwhh_l1f, (int)WHH_E);
  f32_to_bf16_k<<<256, 256, 0, stream>>>(whh_l1b, bwhh_l1b, (int)WHH_E);
  bias_comb_k<<<4, 256, 0, stream>>>(bih_l0f, bhh_l0f, bias_l0f, GDIM);
  bias_comb_k<<<4, 256, 0, stream>>>(bih_l0b, bhh_l0b, bias_l0b, GDIM);
  bias_comb_k<<<4, 256, 0, stream>>>(bih_l1f, bhh_l1f, bias_l1f, GDIM);
  bias_comb_k<<<4, 256, 0, stream>>>(bih_l1b, bhh_l1b, bias_l1b, GDIM);

  const dim3 ggrid(320, 8, 2);

  // 2) layer 0: input GEMM (both dirs) then recurrent scan
  gemm_xg_k<<<ggrid, 256, 0, stream>>>(xbf, bwih_l0f, bwih_l0b,
                                       bias_l0f, bias_l0b, xg0, xg1);
  lstm_rec_k<<<NBLK, 256, 0, stream>>>(xg0, xg1, bwhh_l0f, bwhh_l0b,
                                       hbuf, h0bf, nullptr, bar);

  // 3) layer 1: input GEMM over h0 then recurrent scan -> fp32 d_out
  gemm_xg_k<<<ggrid, 256, 0, stream>>>(h0bf, bwih_l1f, bwih_l1b,
                                       bias_l1f, bias_l1b, xg0, xg1);
  lstm_rec_k<<<NBLK, 256, 0, stream>>>(xg0, xg1, bwhh_l1f, bwhh_l1b,
                                       hbuf, nullptr, (float*)d_out, bar);
}
